// LSTMTSP_2199023255582
// MI455X (gfx1250) — matrix operand order repaired and checked by reading, not compiled, not run on hardware
//
#include <hip/hip_runtime.h>
#include <hip/hip_bf16.h>
#include <math.h>
#include <stdint.h>

// Problem constants (from reference)
constexpr int kB = 256;      // batch
constexpr int kS = 512;      // sequence
constexpr int kE = 256;      // embed
constexpr int kH = 256;      // hidden
constexpr int kG = 4 * kH;   // 1024 gate width
constexpr float kMaskVal = -100000.0f;

typedef __attribute__((ext_vector_type(16))) __bf16 bf16x16;
typedef __attribute__((ext_vector_type(8)))  __bf16 bf16x8;
typedef __attribute__((ext_vector_type(8)))  float  f32x8;

// ---------------------------------------------------------------------------
// A-matrix fragment (16x32 bf16): lane<16 holds K {k0..k0+15},
// lane>=16 holds K {k0+16..k0+31}; both are two contiguous 16B chunks.
__device__ __forceinline__ bf16x16 load_a_frag(const __bf16* __restrict__ row,
                                               int k0, int half) {
  union { bf16x16 v; bf16x8 h[2]; } u;
  u.h[0] = *(const bf16x8*)(row + k0 + half * 16);
  u.h[1] = *(const bf16x8*)(row + k0 + half * 16 + 8);
  return u.v;
}

// async 16B global->LDS copy (ASYNCcnt-tracked, VGLOBAL encoding, no VGPR data)
__device__ __forceinline__ void async_g2l_b128(unsigned lds_off,
                                               unsigned long long gaddr) {
  asm volatile("global_load_async_to_lds_b128 %0, %1, off"
               :: "v"(lds_off), "v"(gaddr) : "memory");
}
__device__ __forceinline__ void wait_asynccnt0() {
  asm volatile("s_wait_asynccnt 0x0" ::: "memory");
}

// ---------------------------------------------------------------------------
// Fused LSTM-gate GEMM: gates[B x 4H] = X[B x E] @ Wih^T + H[B x H] @ Whh^T + bih + bhh
// The 16-row weight tile (shared by all 8 waves of the block) is staged into LDS
// via async global->LDS copies; A rows stream from global per wave.
__global__ __launch_bounds__(256) void gates_gemm_kernel(
    const __hip_bfloat16* __restrict__ Xin, long ldx,
    const __hip_bfloat16* __restrict__ Hin,
    const __hip_bfloat16* __restrict__ WihIn,
    const __hip_bfloat16* __restrict__ WhhIn,
    const float* __restrict__ bih,
    const float* __restrict__ bhh,
    float* __restrict__ gates) {
  // 16 rows x 256 cols bf16, for Wih and Whh tiles (8KB each)
  __shared__ alignas(64) __hip_bfloat16 smW[2 * 16 * kE];

  const __bf16* X  = reinterpret_cast<const __bf16*>(Xin);
  const __bf16* Hs = reinterpret_cast<const __bf16*>(Hin);

  const int wave = threadIdx.x >> 5;
  const int lane = threadIdx.x & 31;
  // tn uniform across the block; tm unique per wave. 128 blocks cover 16x64 tiles.
  const int tn = (blockIdx.x >> 1) << 4;
  const int tm = (((blockIdx.x & 1) << 3) + wave) << 4;
  const int half = lane >> 4;
  const int lrow = lane & 15;

  // ---- stage weight tiles (16 rows each) into LDS asynchronously ----
  {
    const unsigned tid = threadIdx.x;
    const unsigned long long gWih =
        (unsigned long long)(uintptr_t)(WihIn + (size_t)tn * kE);
    const unsigned long long gWhh =
        (unsigned long long)(uintptr_t)(WhhIn + (size_t)tn * kH);
    const unsigned ldsBase = (unsigned)(uintptr_t)(&smW[0]);  // addr[31:0] = LDS offset
    // 16 rows * 512B = 8192B = 512 chunks of 16B per matrix
#pragma unroll
    for (int p = 0; p < 2; ++p) {
      const unsigned j = tid + (unsigned)p * 256u;
      async_g2l_b128(ldsBase + j * 16u, gWih + (unsigned long long)j * 16u);
      async_g2l_b128(ldsBase + 8192u + j * 16u, gWhh + (unsigned long long)j * 16u);
    }
    wait_asynccnt0();
  }
  __syncthreads();

  const __bf16* sWih = reinterpret_cast<const __bf16*>(&smW[0]);
  const __bf16* sWhh = sWih + 16 * kE;

  f32x8 acc = {};

  const __bf16* arow = X + (long)(tm + lrow) * ldx;
  const __bf16* brow = sWih + (size_t)lrow * kE;
#pragma unroll
  for (int k0 = 0; k0 < kE; k0 += 32) {
    bf16x16 a = load_a_frag(arow, k0, half);
    bf16x16 b = *(const bf16x16*)(brow + k0 + half * 16);
    acc = __builtin_amdgcn_wmma_f32_16x16x32_bf16(false, a, false, b,
                                                  (short)0, acc, false, false);
  }

  const __bf16* arow2 = Hs + (size_t)(tm + lrow) * kH;
  const __bf16* brow2 = sWhh + (size_t)lrow * kH;
#pragma unroll
  for (int k0 = 0; k0 < kH; k0 += 32) {
    bf16x16 a = load_a_frag(arow2, k0, half);
    bf16x16 b = *(const bf16x16*)(brow2 + k0 + half * 16);
    acc = __builtin_amdgcn_wmma_f32_16x16x32_bf16(false, a, false, b,
                                                  (short)0, acc, false, false);
  }

  const int col = tn + lrow;
  const float bias = bih[col] + bhh[col];
#pragma unroll
  for (int r = 0; r < 8; ++r) {
    gates[(size_t)(tm + r + 8 * half) * kG + col] = acc[r] + bias;
  }
}

// ---------------------------------------------------------------------------
// Generic H->H GEMM: C[B x H] = A[B x H] @ W[H x H]^T. Output f32 (Cf) or bf16 (Cb),
// row stride ldc (ref_proj is written column-slice-strided into [B,S,H]).
__global__ __launch_bounds__(256) void gemm_hh_kernel(
    const __hip_bfloat16* __restrict__ Ain,
    const __hip_bfloat16* __restrict__ Win,
    float* __restrict__ Cf,
    __hip_bfloat16* __restrict__ Cb,
    long ldc) {
  const __bf16* A = reinterpret_cast<const __bf16*>(Ain);
  const __bf16* W = reinterpret_cast<const __bf16*>(Win);

  const int wave = threadIdx.x >> 5;
  const int lane = threadIdx.x & 31;
  const int tile = blockIdx.x * 8 + wave;   // 16 x 16 = 256 tiles
  const int tm = (tile & 15) << 4;
  const int tn = (tile >> 4) << 4;
  const int half = lane >> 4;
  const int lrow = lane & 15;

  f32x8 acc = {};
  const __bf16* arow = A + (size_t)(tm + lrow) * kH;
  const __bf16* brow = W + (size_t)(tn + lrow) * kH;
#pragma unroll
  for (int k0 = 0; k0 < kH; k0 += 32) {
    bf16x16 a = load_a_frag(arow, k0, half);
    bf16x16 b = *(const bf16x16*)(brow + k0 + half * 16);
    acc = __builtin_amdgcn_wmma_f32_16x16x32_bf16(false, a, false, b,
                                                  (short)0, acc, false, false);
  }
  const int col = tn + lrow;
#pragma unroll
  for (int r = 0; r < 8; ++r) {
    const int row = tm + r + 8 * half;
    if (Cf) Cf[(size_t)row * ldc + col] = acc[r];
    else    Cb[(size_t)row * ldc + col] = __float2bfloat16(acc[r]);
  }
}

// ---------------------------------------------------------------------------
__global__ void f32_to_bf16_kernel(const float* __restrict__ src,
                                   __hip_bfloat16* __restrict__ dst, int n) {
  int i = blockIdx.x * blockDim.x + threadIdx.x;
  if (i < n) dst[i] = __float2bfloat16(src[i]);
}

__global__ void embed_kernel(const float* __restrict__ inputs,
                             const float* __restrict__ W_embed,
                             __hip_bfloat16* __restrict__ emb) {
  long i = (long)blockIdx.x * blockDim.x + threadIdx.x;
  long n = (long)kB * kS * kE;
  if (i >= n) return;
  int e = (int)(i % kE);
  long bs = i / kE;
  float x0 = inputs[bs * 2 + 0];
  float x1 = inputs[bs * 2 + 1];
  emb[i] = __float2bfloat16(x0 * W_embed[e] + x1 * W_embed[kE + e]);
}

__global__ void init_state_kernel(float* h, float* c, __hip_bfloat16* hbf) {
  int i = blockIdx.x * blockDim.x + threadIdx.x;   // B*H
  h[i] = 0.f; c[i] = 0.f; hbf[i] = __float2bfloat16(0.f);
}

__device__ __forceinline__ float sigmoidf_(float x) {
  return 1.0f / (1.0f + __expf(-x));
}

__global__ void lstm_cell_kernel(const float* __restrict__ gates,
                                 float* __restrict__ h, float* __restrict__ c,
                                 __hip_bfloat16* __restrict__ hbf) {
  int i = blockIdx.x * blockDim.x + threadIdx.x;   // B*H
  int b = i >> 8, j = i & (kH - 1);
  const float* g = gates + (size_t)b * kG;
  float ig = sigmoidf_(g[j]);
  float fg = sigmoidf_(g[kH + j]);
  float gg = tanhf(g[2 * kH + j]);
  float og = sigmoidf_(g[3 * kH + j]);
  float cn = fg * c[i] + ig * gg;
  float hn = og * tanhf(cn);
  c[i] = cn; h[i] = hn; hbf[i] = __float2bfloat16(hn);
}

__global__ void dec_init_kernel(unsigned* __restrict__ mask,
                                __hip_bfloat16* __restrict__ xnext,
                                const __hip_bfloat16* __restrict__ emb,
                                float* __restrict__ out_idx) {
  int i = blockIdx.x * blockDim.x + threadIdx.x;   // B*S threads
  if (i < kB * kS) mask[i] = ((i % kS) == 0) ? 1u : 0u;       // START == 0
  if (i < kB * kE) {
    int b = i / kE, e = i % kE;
    xnext[i] = emb[((size_t)b * kS + 0) * kE + e];
  }
  if (i < kB) out_idx[(size_t)i * kS + 0] = 0.0f;
}

// deterministic Gumbel noise from (t,b,s) via splitmix64
__device__ __forceinline__ float gumbelf_(unsigned t, unsigned b, unsigned s) {
  unsigned long long x = ((unsigned long long)(t + 1) << 40) ^
                         ((unsigned long long)(b + 1) << 20) ^
                         (unsigned long long)(s + 0x9e37u);
  x ^= x >> 30; x *= 0xbf58476d1ce4e5b9ULL;
  x ^= x >> 27; x *= 0x94d049bb133111ebULL;
  x ^= x >> 31;
  float u = (float)((x >> 40) + 1ull) * (1.0f / 16777218.0f);  // (0,1)
  return -logf(-logf(u));
}

// ---------------------------------------------------------------------------
// One block per batch row: logits[s] = v . tanh(q[b] + ref_proj[b,s]) (masked),
// then logsumexp + Gumbel-max categorical sample + bookkeeping.
__global__ __launch_bounds__(256) void attn_sample_kernel(
    const float* __restrict__ q,
    const __hip_bfloat16* __restrict__ refp,
    const float* __restrict__ v,
    unsigned* __restrict__ mask,
    const __hip_bfloat16* __restrict__ emb,
    __hip_bfloat16* __restrict__ xnext,
    float* __restrict__ out_lp,
    float* __restrict__ out_idx,
    int t) {
  __shared__ float logits[kS];
  __shared__ float sred[256];
  __shared__ int   sidx[256];
  __shared__ int   chosen_sh;

  const int tid = threadIdx.x;
  const int b = blockIdx.x;
  const int wave = tid >> 5, lane = tid & 31;

  const float* qb = q + (size_t)b * kH;
  float qv[8], vv[8];
#pragma unroll
  for (int r = 0; r < 8; ++r) {
    qv[r] = qb[lane + 32 * r];
    vv[r] = v[lane + 32 * r];
  }
  const __hip_bfloat16* refb = refp + (size_t)b * kS * kH;
  const unsigned* maskb = mask + (size_t)b * kS;

  for (int s = wave; s < kS; s += 8) {       // one wave32 per position
    const __hip_bfloat16* rp = refb + (size_t)s * kH;
    float acc = 0.f;
#pragma unroll
    for (int r = 0; r < 8; ++r)
      acc += vv[r] * tanhf(qv[r] + __bfloat162float(rp[lane + 32 * r]));
#pragma unroll
    for (int off = 16; off > 0; off >>= 1) acc += __shfl_xor(acc, off, 32);
    if (lane == 0) logits[s] = maskb[s] ? kMaskVal : acc;
  }
  __syncthreads();

  float l0 = logits[tid], l1 = logits[tid + 256];
  // block max
  sred[tid] = fmaxf(l0, l1); __syncthreads();
  for (int st = 128; st > 0; st >>= 1) {
    if (tid < st) sred[tid] = fmaxf(sred[tid], sred[tid + st]);
    __syncthreads();
  }
  float maxl = sred[0]; __syncthreads();
  // sum exp
  sred[tid] = __expf(l0 - maxl) + __expf(l1 - maxl); __syncthreads();
  for (int st = 128; st > 0; st >>= 1) {
    if (tid < st) sred[tid] += sred[tid + st];
    __syncthreads();
  }
  float lse = maxl + logf(sred[0]); __syncthreads();
  // Gumbel-max argmax
  float g0 = l0 + gumbelf_(t, b, tid);
  float g1 = l1 + gumbelf_(t, b, tid + 256);
  float bv; int bi;
  if (g0 >= g1) { bv = g0; bi = tid; } else { bv = g1; bi = tid + 256; }
  sred[tid] = bv; sidx[tid] = bi; __syncthreads();
  for (int st = 128; st > 0; st >>= 1) {
    if (tid < st && sred[tid + st] > sred[tid]) {
      sred[tid] = sred[tid + st]; sidx[tid] = sidx[tid + st];
    }
    __syncthreads();
  }
  if (tid == 0) {
    int ch = sidx[0];
    chosen_sh = ch;
    out_lp[(size_t)b * (kS - 1) + t] = logits[ch] - lse;
    out_idx[(size_t)b * kS + t + 1] = (float)ch;
    mask[(size_t)b * kS + ch] = 1u;
  }
  __syncthreads();
  int ch = chosen_sh;
  xnext[(size_t)b * kE + tid] = emb[((size_t)b * kS + ch) * kE + tid];
}

// ---------------------------------------------------------------------------
extern "C" void kernel_launch(void* const* d_in, const int* in_sizes, int n_in,
                              void* d_out, int out_size, void* d_ws, size_t ws_size,
                              hipStream_t stream) {
  const float* inputs   = (const float*)d_in[0];
  const float* W_embed  = (const float*)d_in[1];
  const float* enc_Wih  = (const float*)d_in[2];
  const float* enc_Whh  = (const float*)d_in[3];
  const float* enc_bih  = (const float*)d_in[4];
  const float* enc_bhh  = (const float*)d_in[5];
  const float* dec_Wih  = (const float*)d_in[6];
  const float* dec_Whh  = (const float*)d_in[7];
  const float* dec_bih  = (const float*)d_in[8];
  const float* dec_bhh  = (const float*)d_in[9];
  const float* Wq       = (const float*)d_in[10];
  const float* Wref     = (const float*)d_in[11];
  const float* vvec     = (const float*)d_in[12];

  // bump-allocate workspace (256B aligned)
  char* ws = (char*)d_ws;
  auto alloc = [&](size_t bytes) -> void* {
    void* p = (void*)ws;
    ws += (bytes + 255) & ~(size_t)255;
    return p;
  };
  __hip_bfloat16* encWihB = (__hip_bfloat16*)alloc((size_t)kG * kE * 2);
  __hip_bfloat16* encWhhB = (__hip_bfloat16*)alloc((size_t)kG * kH * 2);
  __hip_bfloat16* decWihB = (__hip_bfloat16*)alloc((size_t)kG * kE * 2);
  __hip_bfloat16* decWhhB = (__hip_bfloat16*)alloc((size_t)kG * kH * 2);
  __hip_bfloat16* WqB     = (__hip_bfloat16*)alloc((size_t)kH * kH * 2);
  __hip_bfloat16* WrefB   = (__hip_bfloat16*)alloc((size_t)kH * kH * 2);
  __hip_bfloat16* embB    = (__hip_bfloat16*)alloc((size_t)kB * kS * kE * 2);
  __hip_bfloat16* refpB   = (__hip_bfloat16*)alloc((size_t)kB * kS * kH * 2);
  float*          hbuf    = (float*)alloc((size_t)kB * kH * 4);
  float*          cbuf    = (float*)alloc((size_t)kB * kH * 4);
  __hip_bfloat16* hbf     = (__hip_bfloat16*)alloc((size_t)kB * kH * 2);
  float*          gates   = (float*)alloc((size_t)kB * kG * 4);
  float*          qf      = (float*)alloc((size_t)kB * kH * 4);
  __hip_bfloat16* xB      = (__hip_bfloat16*)alloc((size_t)kB * kE * 2);
  unsigned*       maskB   = (unsigned*)alloc((size_t)kB * kS * 4);

  // 1) weight precision conversion (once)
  {
    int n1 = kG * kE;
    f32_to_bf16_kernel<<<(n1 + 255) / 256, 256, 0, stream>>>(enc_Wih, encWihB, n1);
    f32_to_bf16_kernel<<<(n1 + 255) / 256, 256, 0, stream>>>(enc_Whh, encWhhB, n1);
    f32_to_bf16_kernel<<<(n1 + 255) / 256, 256, 0, stream>>>(dec_Wih, decWihB, n1);
    f32_to_bf16_kernel<<<(n1 + 255) / 256, 256, 0, stream>>>(dec_Whh, decWhhB, n1);
    int n2 = kH * kH;
    f32_to_bf16_kernel<<<(n2 + 255) / 256, 256, 0, stream>>>(Wq, WqB, n2);
    f32_to_bf16_kernel<<<(n2 + 255) / 256, 256, 0, stream>>>(Wref, WrefB, n2);
  }

  // 2) embedding (once)
  {
    long ne = (long)kB * kS * kE;
    embed_kernel<<<(unsigned)((ne + 255) / 256), 256, 0, stream>>>(inputs, W_embed, embB);
  }

  // 3) encoder scan
  init_state_kernel<<<kB * kH / 256, 256, 0, stream>>>(hbuf, cbuf, hbf);
  for (int t = 0; t < kS; ++t) {
    gates_gemm_kernel<<<128, 256, 0, stream>>>(
        embB + (size_t)t * kE, (long)kS * kE, hbf,
        encWihB, encWhhB, enc_bih, enc_bhh, gates);
    lstm_cell_kernel<<<kB * kH / 256, 256, 0, stream>>>(gates, hbuf, cbuf, hbf);
    // ref_proj[:, t, :] = h @ Wref^T   (bf16, strided into [B,S,H])
    gemm_hh_kernel<<<32, 256, 0, stream>>>(hbf, WrefB, nullptr,
                                           refpB + (size_t)t * kH, (long)kS * kH);
  }

  // 4) decoder scan
  dec_init_kernel<<<(kB * kS + 255) / 256, 256, 0, stream>>>(maskB, xB, embB,
                                                             (float*)d_out + (size_t)kB * (kS - 1));
  float* out_lp  = (float*)d_out;
  float* out_idx = out_lp + (size_t)kB * (kS - 1);
  for (int t = 0; t < kS - 1; ++t) {
    gates_gemm_kernel<<<128, 256, 0, stream>>>(
        xB, (long)kE, hbf, decWihB, decWhhB, dec_bih, dec_bhh, gates);
    lstm_cell_kernel<<<kB * kH / 256, 256, 0, stream>>>(gates, hbuf, cbuf, hbf);
    gemm_hh_kernel<<<32, 256, 0, stream>>>(hbf, WqB, qf, nullptr, (long)kH);
    attn_sample_kernel<<<kB, 256, 0, stream>>>(qf, refpB, vvec, maskB, embB, xB,
                                               out_lp, out_idx, t);
  }
  (void)in_sizes; (void)n_in; (void)out_size; (void)ws_size;
}
